// LIFMCRefracCell_40913858462199
// MI455X (gfx1250) — matrix operand order repaired and checked by reading, not compiled, not run on hardware
//
#include <hip/hip_runtime.h>
#include <hip/hip_bf16.h>

#define BDIM 32768
#define HDIM 256
#define IDIM 128

typedef __attribute__((ext_vector_type(16))) __bf16 v16bf;
typedef __attribute__((ext_vector_type(8)))  float  v8f;

// -------- weight conversion: f32 -> bf16 staged in d_ws --------
// ws layout (bf16 elements): g[H*H] | rec[H*H] | w_in[H*I]
__global__ __launch_bounds__(256)
void cvt_weights_kernel(const float* __restrict__ g,
                        const float* __restrict__ rec,
                        const float* __restrict__ win,
                        unsigned short* __restrict__ ws_raw) {
  __bf16* ws = reinterpret_cast<__bf16*>(ws_raw);
  int idx = blockIdx.x * blockDim.x + threadIdx.x;
  const int NG = HDIM * HDIM;
  const int total = 2 * NG + HDIM * IDIM;
  if (idx >= total) return;
  float val;
  if (idx < NG)            val = g[idx];
  else if (idx < 2 * NG)   val = rec[idx - NG];
  else                     val = win[idx - 2 * NG];
  ws[idx] = (__bf16)val;
}

// A fragment: 16x32 bf16 from f32 row-major [row0..row0+15] x [kk..kk+31]
// ISA layout: lane<16 -> M=lane, regs0-3 = K 8*half..+7, regs4-7 = K 16+8*half..+7
__device__ __forceinline__ v16bf load_a_f32(const float* __restrict__ p, int ld,
                                            int row0, int kk, int m, int half) {
  const float* p0 = p + (size_t)(row0 + m) * ld + kk + 8 * half;
  v16bf a;
#pragma unroll
  for (int j = 0; j < 8; ++j) {
    a[j]     = (__bf16)p0[j];
    a[j + 8] = (__bf16)p0[j + 16];
  }
  return a;
}

// B fragment for X@W.T: B[k][n] = W[col0+n][k]; per lane K 8*half..+7 and 16+8*half..+7
__device__ __forceinline__ v16bf load_b_bf16(const __bf16* __restrict__ w, int ld,
                                             int col0, int kk, int m, int half) {
  const __bf16* q = w + (size_t)(col0 + m) * ld + kk + 8 * half;
  v16bf b;
#pragma unroll
  for (int j = 0; j < 8; ++j) { b[j] = q[j]; b[j + 8] = q[j + 16]; }
  return b;
}

__device__ __forceinline__ v8f wmma_bf16(v16bf a, v16bf b, v8f c) {
  return __builtin_amdgcn_wmma_f32_16x16x32_bf16(false, a, false, b, (short)0, c,
                                                 false, false);
}

// -------- fused LIF-MC refractory step --------
// block = 256 threads (8 waves), tile = 64 rows x 64 cols of (B,H)
// wave w: rowTile = w>>1, colPair = w&1 -> cols [n0, n0+32)
__global__ __launch_bounds__(256)
void lif_mc_refrac_kernel(const float* __restrict__ inp,
                          const float* __restrict__ z,
                          const float* __restrict__ v,
                          const float* __restrict__ i_in,
                          const float* __restrict__ rho,
                          const unsigned short* __restrict__ ws_raw,
                          float* __restrict__ out) {
  const __bf16* gBF   = reinterpret_cast<const __bf16*>(ws_raw);
  const __bf16* recBF = gBF + HDIM * HDIM;
  const __bf16* winBF = gBF + 2 * HDIM * HDIM;

  const int lane  = threadIdx.x & 31;
  const int wave  = threadIdx.x >> 5;
  const int m     = lane & 15;
  const int half  = lane >> 4;
  const int rBase = blockIdx.x * 64 + (wave >> 1) * 16;
  const int n0    = blockIdx.y * 64 + (wave & 1) * 32;

  v8f accV0 = {}, accV1 = {};   // v @ g_coupling.T
  v8f accI0 = {}, accI1 = {};   // z @ rec.T + input @ w_in.T

  // K over H for the two HxH GEMMs
#pragma unroll 2
  for (int kk = 0; kk < HDIM; kk += 32) {
    v16bf av  = load_a_f32(v, HDIM, rBase, kk, m, half);
    v16bf az  = load_a_f32(z, HDIM, rBase, kk, m, half);
    v16bf bg0 = load_b_bf16(gBF,   HDIM, n0,      kk, m, half);
    v16bf bg1 = load_b_bf16(gBF,   HDIM, n0 + 16, kk, m, half);
    v16bf br0 = load_b_bf16(recBF, HDIM, n0,      kk, m, half);
    v16bf br1 = load_b_bf16(recBF, HDIM, n0 + 16, kk, m, half);
    accV0 = wmma_bf16(av, bg0, accV0);
    accV1 = wmma_bf16(av, bg1, accV1);
    accI0 = wmma_bf16(az, br0, accI0);
    accI1 = wmma_bf16(az, br1, accI1);
  }
  // K over I for the input GEMM
#pragma unroll 2
  for (int kk = 0; kk < IDIM; kk += 32) {
    v16bf ax  = load_a_f32(inp, IDIM, rBase, kk, m, half);
    v16bf bw0 = load_b_bf16(winBF, IDIM, n0,      kk, m, half);
    v16bf bw1 = load_b_bf16(winBF, IDIM, n0 + 16, kk, m, half);
    accI0 = wmma_bf16(ax, bw0, accI0);
    accI1 = wmma_bf16(ax, bw1, accI1);
  }

  // Epilogue. C/D layout: lane<16 -> N=lane, M=reg; lane>=16 -> M=reg+8
  // i_in / rho are single-use -> nontemporal loads; outputs are write-once,
  // never re-read on device -> nontemporal stores (don't thrash 192MB L2).
  const size_t BH = (size_t)BDIM * HDIM;
  const int grBase = rBase + half * 8;
#pragma unroll
  for (int t = 0; t < 2; ++t) {
    v8f vg = t ? accV1 : accV0;
    v8f gi = t ? accI1 : accI0;
    const int gc = n0 + t * 16 + m;
#pragma unroll
    for (int r = 0; r < 8; ++r) {
      const size_t off = (size_t)(grBase + r) * HDIM + gc;
      const float vv = v[off];                               // L2-hot (A reuse)
      const float ci = __builtin_nontemporal_load(i_in + off);
      const float rr = __builtin_nontemporal_load(rho + off);
      // v_decayed = v + 0.1*((0 - v) + i) + (v@g.T)
      const float v_dec = 0.9f * vv + 0.1f * ci + vg[r];
      // spike
      float zf = (v_dec - 1.0f > 0.0f) ? 1.0f : 0.0f;
      float vn = (1.0f - zf) * v_dec;                 // V_RESET == 0
      const float in_new = 0.8f * ci + gi[r];         // i_decayed + gemms
      // refractory
      const float mk = (rr > 0.0f) ? 1.0f : 0.0f;
      vn = (1.0f - mk) * vn + mk * vv;
      zf = (1.0f - mk) * zf;
      const float rho_n = (1.0f - zf) * fmaxf(rr - mk, 0.0f) + zf * 5.0f;

      __builtin_nontemporal_store(zf,     out + off);
      __builtin_nontemporal_store(vn,     out + BH + off);
      __builtin_nontemporal_store(in_new, out + 2 * BH + off);
      __builtin_nontemporal_store(rho_n,  out + 3 * BH + off);
    }
  }
}

extern "C" void kernel_launch(void* const* d_in, const int* in_sizes, int n_in,
                              void* d_out, int out_size, void* d_ws, size_t ws_size,
                              hipStream_t stream) {
  const float* inp = (const float*)d_in[0];  // (B, I)
  const float* z   = (const float*)d_in[1];  // (B, H)
  const float* v   = (const float*)d_in[2];  // (B, H)
  const float* ii  = (const float*)d_in[3];  // (B, H)
  const float* rho = (const float*)d_in[4];  // (B, H)
  const float* win = (const float*)d_in[5];  // (H, I)
  const float* rec = (const float*)d_in[6];  // (H, H)
  const float* g   = (const float*)d_in[7];  // (H, H)

  unsigned short* wbf = (unsigned short*)d_ws;

  const int total = 2 * HDIM * HDIM + HDIM * IDIM;
  cvt_weights_kernel<<<(total + 255) / 256, 256, 0, stream>>>(g, rec, win, wbf);

  dim3 grid(BDIM / 64, HDIM / 64);
  lif_mc_refrac_kernel<<<grid, dim3(256), 0, stream>>>(inp, z, v, ii, rho, wbf,
                                                       (float*)d_out);
}
